// CascadeRoIHeads_2396591751882
// MI455X (gfx1250) — matrix operand order repaired and checked by reading, not compiled, not run on hardware
//
#include <hip/hip_runtime.h>

// ---------------------------------------------------------------------------
// Cascade RoI heads for MI455X (gfx1250):
//   bf16 WMMA GEMMs fed by the Tensor Data Mover (tensor_load_to_lds),
//   VALU glue for RoIAlign / decode / softmax / NMS.
// ---------------------------------------------------------------------------

#define BATCH      2
#define NPROP      512
#define CFEAT      256
#define FH         200
#define FW         200
#define NCLS       81
#define NSTAGES    3
#define POOL       7
#define SRR        2
#define ROI_SCALE  0.25f
#define IMGH       800.0f
#define IMGW       800.0f
#define FCDIM      1024
#define INDIM      (CFEAT * POOL * POOL)   // 12544
#define MTOT       (BATCH * NPROP)         // 1024
#define CLIPV      4.135166556742356f      // log(1000/16)
#define SCORE_T    0.05f
#define NMS_T      0.5f
#define DETS       100

typedef __attribute__((ext_vector_type(16))) __bf16       v16bf;
typedef __attribute__((ext_vector_type(8)))  float        v8f;
typedef __attribute__((ext_vector_type(4)))  unsigned int u32x4;
typedef __attribute__((ext_vector_type(8)))  int          i32x8;
typedef __attribute__((ext_vector_type(4)))  int          i32x4;

struct B32x { uint4 lo, hi; };   // 32 bytes == v16bf

// ---------------------------------------------------------------------------
// TDM: DMA one [128 rows x 32 elems] bf16 2D tile (row stride = K elems) from
// global into LDS at lds_off, inserting 4 DWORDs of pad after every 16 DWORDs
// so the LDS row stride becomes 40 bf16 (conflict-free for fragment loads).
// Descriptor bitfields per CDNA5 ISA.
// ---------------------------------------------------------------------------
__device__ __forceinline__ void tdm_load_tile_bf16(unsigned int lds_off,
                                                   const __bf16* gptr,
                                                   unsigned int row_len_elems,
                                                   unsigned int rows_total) {
  const unsigned long long ga = (unsigned long long)(uintptr_t)gptr;

  u32x4 g0;
  g0.x = 1u;                                   // count=1, user mode
  g0.y = lds_off;                              // lds_addr [63:32]
  g0.z = (unsigned int)ga;                     // global_addr[31:0]
  g0.w = (unsigned int)((ga >> 32) & 0x1FFFFFFu) | (2u << 30);  // [120:96] + type=2

  const unsigned int td0 = row_len_elems;      // tensor_dim0 (elems)
  const unsigned int td1 = rows_total;         // tensor_dim1 (rows)
  const unsigned long long st0 = (unsigned long long)row_len_elems;

  i32x8 g1 = {
      // mask=0 | data_size=1(2B)<<16 | pad_en<<20 | pad_interval=3(16dw)<<22
      // | pad_amount=3(4dw)<<25
      (int)((1u << 16) | (1u << 20) | (3u << 22) | (3u << 25)),
      (int)((td0 & 0xFFFFu) << 16),                       // [63:48]=td0.lo16
      (int)((td0 >> 16) | ((td1 & 0xFFFFu) << 16)),       // td0.hi16 | td1.lo16
      (int)((td1 >> 16) | (32u << 16)),                   // td1.hi16 | tile_dim0=32
      (int)(128u),                                        // tile_dim1=128, tile_dim2=0
      (int)(unsigned int)(st0 & 0xFFFFFFFFu),             // dim0_stride[31:0]
      (int)(unsigned int)((st0 >> 32) & 0xFFFFu),         // dim0_stride[47:32]
      0};                                                 // dim1_stride (unused, 2D)

  const i32x4 gz = {0, 0, 0, 0};
#if defined(__clang_major__) && (__clang_major__ >= 23)
  const i32x8 gz8 = {0, 0, 0, 0, 0, 0, 0, 0};
  __builtin_amdgcn_tensor_load_to_lds(g0, g1, gz, gz, gz8, 0);
#else
  __builtin_amdgcn_tensor_load_to_lds(g0, g1, gz, gz, 0);
#endif
}

// ---- WMMA fragment gathers from LDS (layouts per CDNA5 ISA 7.12.2) --------
// A (16x32 bf16): lane<16 -> row=lane,    K = {kh..kh+7, kh+16..kh+23}, kh=0
//                 lane>=16 -> row=lane-16, same with kh=8
__device__ __forceinline__ v16bf lds_frag_a(const __bf16* As, int mbase, int lane) {
  const int row = mbase + (lane & 15);
  const int kh  = (lane >> 4) * 8;
  const __bf16* p = As + row * 40 + kh;
  B32x r;
  r.lo = *(const uint4*)(p);
  r.hi = *(const uint4*)(p + 16);
  return __builtin_bit_cast(v16bf, r);
}
// B (32x16 bf16), stored in LDS as [N][K] (weights pre-transposed):
// lane holds column N=lane&15, K = {kh..kh+7, kh+16..kh+23}, kh=(lane>>4)*8
__device__ __forceinline__ v16bf lds_frag_b(const __bf16* Bs, int nbase, int lane) {
  const int col = nbase + (lane & 15);
  const int kh  = (lane >> 4) * 8;
  const __bf16* p = Bs + col * 40 + kh;
  B32x r;
  r.lo = *(const uint4*)(p);
  r.hi = *(const uint4*)(p + 16);
  return __builtin_bit_cast(v16bf, r);
}

// ---------------------------------------------------------------------------
// GEMM: Out = relu(A[M,K](bf16) * Bt[N,K](bf16)^T + bias[N]),  Out bf16.
// Block tile 128x128, BK=32, 8 waves (2x4), wave tile 64x32 = 4x2 WMMA.
// Both tiles are streamed global->LDS by the Tensor Data Mover (wave 0),
// double-buffered; compute overlaps the next tile's DMA.
// ---------------------------------------------------------------------------
__global__ __launch_bounds__(256) void gemm_bias_relu_wmma_tdm(
    const __bf16* __restrict__ A, const __bf16* __restrict__ Bt,
    const float* __restrict__ bias, __bf16* __restrict__ Out,
    int M, int N, int K) {
  (void)M;
  __shared__ __align__(16) __bf16 As[2][128][40];
  __shared__ __align__(16) __bf16 Bs[2][128][40];

  const int tid  = threadIdx.x;
  const int lane = tid & 31;
  const int wid  = tid >> 5;
  const int wm   = wid >> 2;  // 0..1
  const int wn   = wid & 3;   // 0..3
  const int m0   = blockIdx.y * 128;
  const int n0   = blockIdx.x * 128;

  const unsigned int as_base = (unsigned int)(uintptr_t)&As[0][0][0];
  const unsigned int bs_base = (unsigned int)(uintptr_t)&Bs[0][0][0];
  const unsigned int buf_bytes = 128u * 40u * 2u;  // 10240 B per buffer

  v8f acc[4][2] = {};

  // Prologue: DMA tile 0 into buffer 0.
  if (wid == 0) {
    tdm_load_tile_bf16(as_base, A + (size_t)m0 * K, K, 128);
    tdm_load_tile_bf16(bs_base, Bt + (size_t)n0 * K, K, 128);
    __builtin_amdgcn_s_wait_tensorcnt(0);
  }
  __syncthreads();

  const int nk = K >> 5;
  for (int kt = 0; kt < nk; ++kt) {
    const int buf = kt & 1;

    // Kick off next tile's DMA while this tile computes.
    if (kt + 1 < nk && wid == 0) {
      const size_t ke = (size_t)(kt + 1) << 5;
      tdm_load_tile_bf16(as_base + (buf ^ 1) * buf_bytes, A + (size_t)m0 * K + ke, K, 128);
      tdm_load_tile_bf16(bs_base + (buf ^ 1) * buf_bytes, Bt + (size_t)n0 * K + ke, K, 128);
    }

    v16bf b0 = lds_frag_b(&Bs[buf][0][0], wn * 32 + 0,  lane);
    v16bf b1 = lds_frag_b(&Bs[buf][0][0], wn * 32 + 16, lane);
#pragma unroll
    for (int mt = 0; mt < 4; ++mt) {
      v16bf av = lds_frag_a(&As[buf][0][0], wm * 64 + mt * 16, lane);
      acc[mt][0] = __builtin_amdgcn_wmma_f32_16x16x32_bf16(
          false, av, false, b0, (short)0, acc[mt][0], false, false);
      acc[mt][1] = __builtin_amdgcn_wmma_f32_16x16x32_bf16(
          false, av, false, b1, (short)0, acc[mt][1], false, false);
    }

    if (wid == 0 && kt + 1 < nk) __builtin_amdgcn_s_wait_tensorcnt(0);
    __syncthreads();
  }

  // Epilogue: C layout -> VGPR i: M = i + 8*(lane>=16), N = lane&15.
  const int rbase = m0 + wm * 64;
  const int cbase = n0 + wn * 32 + (lane & 15);
  const int rofs  = (lane >> 4) * 8;
#pragma unroll
  for (int mt = 0; mt < 4; ++mt) {
#pragma unroll
    for (int nt = 0; nt < 2; ++nt) {
      const int cc = cbase + nt * 16;
      const float bv = bias[cc];
#pragma unroll
      for (int i = 0; i < 8; ++i) {
        const int rr = rbase + mt * 16 + rofs + i;
        float v = acc[mt][nt][i] + bv;
        v = v > 0.f ? v : 0.f;
        Out[(size_t)rr * N + cc] = (__bf16)v;
      }
    }
  }
}

// ---------------------------------------------------------------------------
// Weight transpose + f32->bf16: Wt[N][K] = (bf16)W[K][N]. 64x64 LDS tiles.
// ---------------------------------------------------------------------------
__global__ __launch_bounds__(256) void transpose_cvt_kernel(
    const float* __restrict__ W, __bf16* __restrict__ Wt, int K, int N) {
  __shared__ float t[64][65];
  const int tid = threadIdx.x;
  const int tx = tid & 63;
  const int ty = tid >> 6;
  const int k0 = blockIdx.y * 64;
  const int n0 = blockIdx.x * 64;
#pragma unroll
  for (int r = ty; r < 64; r += 4)
    t[r][tx] = W[(size_t)(k0 + r) * N + n0 + tx];
  __syncthreads();
#pragma unroll
  for (int r = ty; r < 64; r += 4)
    Wt[(size_t)(n0 + r) * K + k0 + tx] = (__bf16)t[tx][r];
}

// ---------------------------------------------------------------------------
// RoIAlign: one block per RoI, thread = channel. Writes X[r][c*49 + cell] bf16
// (49 contiguous bf16 per thread).
// ---------------------------------------------------------------------------
__global__ __launch_bounds__(256) void roi_align_kernel(
    const float* __restrict__ feat, const float* __restrict__ rois,
    __bf16* __restrict__ X) {
  const int r = blockIdx.x;       // 0..1023
  const int c = threadIdx.x;      // channel 0..255
  const int b = r >> 9;

  const float* box = rois + (size_t)r * 4;
  const float x1 = box[0] * ROI_SCALE, y1 = box[1] * ROI_SCALE;
  const float x2 = box[2] * ROI_SCALE, y2 = box[3] * ROI_SCALE;
  const float rw = fmaxf(x2 - x1, 1.f), rh = fmaxf(y2 - y1, 1.f);
  const float bw = rw * (1.f / POOL), bh = rh * (1.f / POOL);

  const float* f = feat + ((size_t)b * CFEAT + c) * (FH * FW);
  __bf16* xout = X + (size_t)r * INDIM + c * (POOL * POOL);

  for (int py = 0; py < POOL; ++py) {
    for (int px = 0; px < POOL; ++px) {
      float acc = 0.f;
#pragma unroll
      for (int sy = 0; sy < SRR; ++sy) {
        const float y = y1 + ((float)py + ((float)sy + 0.5f) / SRR) * bh;
#pragma unroll
        for (int sx = 0; sx < SRR; ++sx) {
          const float x = x1 + ((float)px + ((float)sx + 0.5f) / SRR) * bw;
          const float valid =
              (y > -1.f && y < (float)FH && x > -1.f && x < (float)FW) ? 1.f : 0.f;
          const float yc = fminf(fmaxf(y, 0.f), (float)FH - 1.f);
          const float xc = fminf(fmaxf(x, 0.f), (float)FW - 1.f);
          const int y0 = (int)floorf(yc), x0 = (int)floorf(xc);
          const int y1i = (y0 + 1 < FH) ? y0 + 1 : FH - 1;
          const int x1i = (x0 + 1 < FW) ? x0 + 1 : FW - 1;
          const float ly = yc - (float)y0, lx = xc - (float)x0;
          const float hy = 1.f - ly, hx = 1.f - lx;
          const float v = hy * hx * f[y0 * FW + x0] + hy * lx * f[y0 * FW + x1i] +
                          ly * hx * f[y1i * FW + x0] + ly * lx * f[y1i * FW + x1i];
          acc += v * valid;
        }
      }
      xout[py * POOL + px] = (__bf16)(acc * (1.f / (SRR * SRR)));
    }
  }
}

// ---------------------------------------------------------------------------
// Heads: logits = H*Wc + bc (81), deltas = H*Wr + br (4). One block per row.
// ---------------------------------------------------------------------------
__global__ __launch_bounds__(128) void heads_kernel(
    const __bf16* __restrict__ H, const float* __restrict__ Wc,
    const float* __restrict__ bc, const float* __restrict__ Wr,
    const float* __restrict__ br, float* __restrict__ logits,
    float* __restrict__ deltas) {
  __shared__ float hrow[FCDIM];
  const int r = blockIdx.x;
  const int tid = threadIdx.x;
  for (int k = tid; k < FCDIM; k += 128)
    hrow[k] = (float)H[(size_t)r * FCDIM + k];
  __syncthreads();

  if (tid < NCLS) {
    float acc = bc[tid];
    for (int k = 0; k < FCDIM; ++k) acc += hrow[k] * Wc[(size_t)k * NCLS + tid];
    logits[(size_t)r * NCLS + tid] = acc;
  } else if (tid < NCLS + 4) {
    const int j = tid - NCLS;
    float acc = br[j];
    for (int k = 0; k < FCDIM; ++k) acc += hrow[k] * Wr[(size_t)k * 4 + j];
    deltas[(size_t)r * 4 + j] = acc;
  }
}

// ---------------------------------------------------------------------------
// decode_clip (in place on cur boxes)
// ---------------------------------------------------------------------------
__global__ __launch_bounds__(256) void decode_kernel(
    float* __restrict__ cur, const float* __restrict__ deltas,
    float wx, float wy, float ww, float wh) {
  const int i = blockIdx.x * blockDim.x + threadIdx.x;
  if (i >= MTOT) return;
  const float px1 = cur[i * 4 + 0], py1 = cur[i * 4 + 1];
  const float px2 = cur[i * 4 + 2], py2 = cur[i * 4 + 3];
  const float pw = fmaxf(px2 - px1, 1e-6f), ph = fmaxf(py2 - py1, 1e-6f);
  const float px = px1 + 0.5f * pw, py = py1 + 0.5f * ph;
  const float dx = deltas[i * 4 + 0] * wx;
  const float dy = deltas[i * 4 + 1] * wy;
  const float dw = fminf(deltas[i * 4 + 2] * ww, CLIPV);
  const float dh = fminf(deltas[i * 4 + 3] * wh, CLIPV);
  const float gx = dx * pw + px, gy = dy * ph + py;
  const float gw = expf(dw) * pw, gh = expf(dh) * ph;
  cur[i * 4 + 0] = fminf(fmaxf(gx - 0.5f * gw, 0.f), IMGW);
  cur[i * 4 + 1] = fminf(fmaxf(gy - 0.5f * gh, 0.f), IMGH);
  cur[i * 4 + 2] = fminf(fmaxf(gx + 0.5f * gw, 0.f), IMGW);
  cur[i * 4 + 3] = fminf(fmaxf(gy + 0.5f * gh, 0.f), IMGH);
}

__global__ __launch_bounds__(256) void softmax_kernel(
    const float* __restrict__ logits, float* __restrict__ probs) {
  const int r = blockIdx.x * blockDim.x + threadIdx.x;
  if (r >= MTOT) return;
  const float* L = logits + (size_t)r * NCLS;
  float m = L[0];
  for (int i = 1; i < NCLS; ++i) m = fmaxf(m, L[i]);
  float s = 0.f;
  for (int i = 0; i < NCLS; ++i) s += expf(L[i] - m);
  const float inv = 1.f / s;
  for (int i = 0; i < NCLS; ++i) probs[(size_t)r * NCLS + i] = expf(L[i] - m) * inv;
}

// ---------------------------------------------------------------------------
// Greedy class-aware NMS (== reference's offset-box suppression order).
// One block per batch. Writes boxes/scores/labels into d_out.
// ---------------------------------------------------------------------------
__global__ __launch_bounds__(256) void postprocess_kernel(
    const float* __restrict__ boxes, const float* __restrict__ probs,
    float* __restrict__ sc_ws, float* __restrict__ out) {
  const int b = blockIdx.x;
  const int tid = threadIdx.x;
  const int NC = NPROP * (NCLS - 1);
  float* sc = sc_ws + (size_t)b * NC;

  for (int i = tid; i < NC; i += 256) {
    const int n = i / (NCLS - 1);
    const int cls = i - n * (NCLS - 1);
    const float* bx = boxes + ((size_t)b * NPROP + n) * 4;
    const bool ok = (bx[2] - bx[0] >= 1.f) && (bx[3] - bx[1] >= 1.f);
    const float pv = probs[((size_t)b * NPROP + n) * NCLS + cls + 1];
    sc[i] = (ok && pv > SCORE_T) ? pv : -1.f;
  }
  __syncthreads();

  __shared__ float sval[256];
  __shared__ int sidx[256];
  __shared__ float sbox[4];

  float* ob  = out + (size_t)b * DETS * 4;
  float* osc = out + (size_t)BATCH * DETS * 4 + (size_t)b * DETS;
  int* olb = (int*)(out + (size_t)BATCH * DETS * 4 + (size_t)BATCH * DETS) +
             (size_t)b * DETS;

  for (int det = 0; det < DETS; ++det) {
    float bv = -1.f;
    int bi = -1;
    for (int i = tid; i < NC; i += 256) {
      const float v = sc[i];
      if (v > bv) { bv = v; bi = i; }
    }
    sval[tid] = bv;
    sidx[tid] = bi;
    __syncthreads();
    for (int s = 128; s > 0; s >>= 1) {
      if (tid < s && sval[tid + s] > sval[tid]) {
        sval[tid] = sval[tid + s];
        sidx[tid] = sidx[tid + s];
      }
      __syncthreads();
    }
    const float best = sval[0];
    const int bidx = sidx[0];

    if (best <= 0.f) {
      if (tid == 0) {
        ob[det * 4 + 0] = 0.f; ob[det * 4 + 1] = 0.f;
        ob[det * 4 + 2] = 0.f; ob[det * 4 + 3] = 0.f;
        osc[det] = 0.f; olb[det] = 0;
      }
      __syncthreads();
      continue;
    }
    const int bn = bidx / (NCLS - 1);
    const int bcls = bidx - bn * (NCLS - 1);
    if (tid == 0) {
      const float* bb = boxes + ((size_t)b * NPROP + bn) * 4;
      sbox[0] = bb[0]; sbox[1] = bb[1]; sbox[2] = bb[2]; sbox[3] = bb[3];
      ob[det * 4 + 0] = bb[0]; ob[det * 4 + 1] = bb[1];
      ob[det * 4 + 2] = bb[2]; ob[det * 4 + 3] = bb[3];
      osc[det] = best;
      olb[det] = bcls + 1;
      sc[bidx] = -1.f;
    }
    __syncthreads();
    const float bx1 = sbox[0], by1 = sbox[1], bx2 = sbox[2], by2 = sbox[3];
    const float barea = (bx2 - bx1) * (by2 - by1);
    for (int i = tid; i < NC; i += 256) {
      if (sc[i] <= 0.f) continue;
      const int n = i / (NCLS - 1);
      const int cls = i - n * (NCLS - 1);
      if (cls != bcls) continue;  // offset-by-label => same class only
      const float* cb = boxes + ((size_t)b * NPROP + n) * 4;
      const float xx1 = fmaxf(bx1, cb[0]);
      const float yy1 = fmaxf(by1, cb[1]);
      const float xx2 = fminf(bx2, cb[2]);
      const float yy2 = fminf(by2, cb[3]);
      const float w = fmaxf(xx2 - xx1, 0.f), h = fmaxf(yy2 - yy1, 0.f);
      const float inter = w * h;
      const float area = (cb[2] - cb[0]) * (cb[3] - cb[1]);
      const float iou = inter / (barea + area - inter + 1e-12f);
      if (iou > NMS_T) sc[i] = -1.f;
    }
    __syncthreads();
  }
}

__global__ __launch_bounds__(256) void copy_kernel(const float* __restrict__ src,
                                                   float* __restrict__ dst, int n) {
  const int i = blockIdx.x * blockDim.x + threadIdx.x;
  if (i < n) dst[i] = src[i];
}

// ---------------------------------------------------------------------------
extern "C" void kernel_launch(void* const* d_in, const int* in_sizes, int n_in,
                              void* d_out, int out_size, void* d_ws, size_t ws_size,
                              hipStream_t stream) {
  (void)in_sizes; (void)n_in; (void)out_size; (void)ws_size;

  const float* features  = (const float*)d_in[0];
  const float* proposals = (const float*)d_in[1];
  const float* W1 = (const float*)d_in[2];
  const float* b1 = (const float*)d_in[3];
  const float* W2 = (const float*)d_in[4];
  const float* b2 = (const float*)d_in[5];
  const float* Wc = (const float*)d_in[6];
  const float* bc = (const float*)d_in[7];
  const float* Wr = (const float*)d_in[8];
  const float* br = (const float*)d_in[9];

  auto up = [](size_t x) { return (x + 255) & ~(size_t)255; };
  char* p = (char*)d_ws;
  float*  cur    = (float*)p;  p += up(sizeof(float) * MTOT * 4);
  __bf16* X      = (__bf16*)p; p += up(sizeof(__bf16) * (size_t)MTOT * INDIM);
  __bf16* H1     = (__bf16*)p; p += up(sizeof(__bf16) * (size_t)MTOT * FCDIM);
  __bf16* H2     = (__bf16*)p; p += up(sizeof(__bf16) * (size_t)MTOT * FCDIM);
  __bf16* W1t    = (__bf16*)p; p += up(sizeof(__bf16) * (size_t)FCDIM * INDIM);
  __bf16* W2t    = (__bf16*)p; p += up(sizeof(__bf16) * (size_t)FCDIM * FCDIM);
  float*  logits = (float*)p;  p += up(sizeof(float) * (size_t)MTOT * NCLS);
  float*  deltas = (float*)p;  p += up(sizeof(float) * MTOT * 4);
  float*  probs  = (float*)p;  p += up(sizeof(float) * (size_t)MTOT * NCLS);
  float*  sc_ws  = (float*)p;  p += up(sizeof(float) * (size_t)BATCH * NPROP * (NCLS - 1));

  copy_kernel<<<(MTOT * 4 + 255) / 256, 256, 0, stream>>>(proposals, cur, MTOT * 4);

  static const float ST[NSTAGES][4] = {
      {0.1f, 0.1f, 0.2f, 0.2f},
      {0.05f, 0.05f, 0.1f, 0.1f},
      {0.033f, 0.033f, 0.067f, 0.067f}};

  for (int s = 0; s < NSTAGES; ++s) {
    // Pre-transpose + convert this stage's FC weights to bf16 [N][K].
    transpose_cvt_kernel<<<dim3(FCDIM / 64, INDIM / 64), 256, 0, stream>>>(
        W1 + (size_t)s * INDIM * FCDIM, W1t, INDIM, FCDIM);
    transpose_cvt_kernel<<<dim3(FCDIM / 64, FCDIM / 64), 256, 0, stream>>>(
        W2 + (size_t)s * FCDIM * FCDIM, W2t, FCDIM, FCDIM);

    roi_align_kernel<<<dim3(MTOT), 256, 0, stream>>>(features, cur, X);
    gemm_bias_relu_wmma_tdm<<<dim3(FCDIM / 128, MTOT / 128), 256, 0, stream>>>(
        X, W1t, b1 + (size_t)s * FCDIM, H1, MTOT, FCDIM, INDIM);
    gemm_bias_relu_wmma_tdm<<<dim3(FCDIM / 128, MTOT / 128), 256, 0, stream>>>(
        H1, W2t, b2 + (size_t)s * FCDIM, H2, MTOT, FCDIM, FCDIM);
    heads_kernel<<<MTOT, 128, 0, stream>>>(
        H2, Wc + (size_t)s * FCDIM * NCLS, bc + (size_t)s * NCLS,
        Wr + (size_t)s * FCDIM * 4, br + (size_t)s * 4, logits, deltas);
    decode_kernel<<<(MTOT + 255) / 256, 256, 0, stream>>>(
        cur, deltas, ST[s][0], ST[s][1], ST[s][2], ST[s][3]);
  }

  softmax_kernel<<<(MTOT + 255) / 256, 256, 0, stream>>>(logits, probs);
  postprocess_kernel<<<BATCH, 256, 0, stream>>>(cur, probs, sc_ws, (float*)d_out);
}
